// TripleInfoNCELoss_78563541778735
// MI455X (gfx1250) — matrix operand order repaired and checked by reading, not compiled, not run on hardware
//
#include <hip/hip_runtime.h>
#include <hip/hip_bf16.h>

typedef __attribute__((ext_vector_type(16))) _Float16 v16h;
typedef __attribute__((ext_vector_type(8)))  _Float16 v8h;
typedef __attribute__((ext_vector_type(8)))  float    v8f;

#define BQ 8192
#define DQ 256
#define INV_T (1.0f / 0.07f)
// log2(e)/T : one fused scale so exp(sim/T) == v_exp_f32(sim * EXP2_SCALE)
#define EXP2_SCALE (1.4426950408889634f / 0.07f)
#define NSPLIT 8
#define NCOLS_PER_SPLIT (BQ / NSPLIT)   // 1024
#define SB_STRIDE_H 264                 // 256 + 8 halves pad: 528 B rows (16B aligned, banks spread)
#define SB_STRIDE_U32 (SB_STRIDE_H / 2) // 132
#define TILE_HALVES (64 * SB_STRIDE_H)
#define TILE_BYTES  (TILE_HALVES * 2)   // 33792

union Frag16 { unsigned u[8]; v16h h; };

// ---------------------------------------------------------------------------
// Kernel A: q = normalize(h + r) -> f16 ; t = normalize(t) -> f16
// One wave (32 lanes) per row; 8 floats per lane.
// ---------------------------------------------------------------------------
__global__ __launch_bounds__(256) void infonce_normalize(
    const float* __restrict__ h, const float* __restrict__ r,
    const float* __restrict__ t,
    _Float16* __restrict__ qh, _Float16* __restrict__ th)
{
    const int w = threadIdx.x >> 5;
    const int l = threadIdx.x & 31;
    const int row = blockIdx.x * 8 + w;

    const float4* h4 = (const float4*)(h + (size_t)row * DQ);
    const float4* r4 = (const float4*)(r + (size_t)row * DQ);
    const float4* t4 = (const float4*)(t + (size_t)row * DQ);

    // ---- q = normalize(h + r) ----
    float4 xa = h4[l * 2 + 0], xb = h4[l * 2 + 1];
    float4 ya = r4[l * 2 + 0], yb = r4[l * 2 + 1];
    xa.x += ya.x; xa.y += ya.y; xa.z += ya.z; xa.w += ya.w;
    xb.x += yb.x; xb.y += yb.y; xb.z += yb.z; xb.w += yb.w;
    float ss = xa.x*xa.x + xa.y*xa.y + xa.z*xa.z + xa.w*xa.w
             + xb.x*xb.x + xb.y*xb.y + xb.z*xb.z + xb.w*xb.w;
    ss += __shfl_xor(ss, 1, 32); ss += __shfl_xor(ss, 2, 32);
    ss += __shfl_xor(ss, 4, 32); ss += __shfl_xor(ss, 8, 32);
    ss += __shfl_xor(ss, 16, 32);
    float sc = 1.0f / fmaxf(sqrtf(ss), 1e-12f);
    {
        v8h o;
        o[0]=(_Float16)(xa.x*sc); o[1]=(_Float16)(xa.y*sc);
        o[2]=(_Float16)(xa.z*sc); o[3]=(_Float16)(xa.w*sc);
        o[4]=(_Float16)(xb.x*sc); o[5]=(_Float16)(xb.y*sc);
        o[6]=(_Float16)(xb.z*sc); o[7]=(_Float16)(xb.w*sc);
        *(v8h*)(qh + (size_t)row * DQ + l * 8) = o;
    }

    // ---- t = normalize(t) ----
    float4 ta = t4[l * 2 + 0], tb = t4[l * 2 + 1];
    float st = ta.x*ta.x + ta.y*ta.y + ta.z*ta.z + ta.w*ta.w
             + tb.x*tb.x + tb.y*tb.y + tb.z*tb.z + tb.w*tb.w;
    st += __shfl_xor(st, 1, 32); st += __shfl_xor(st, 2, 32);
    st += __shfl_xor(st, 4, 32); st += __shfl_xor(st, 8, 32);
    st += __shfl_xor(st, 16, 32);
    float sct = 1.0f / fmaxf(sqrtf(st), 1e-12f);
    {
        v8h o;
        o[0]=(_Float16)(ta.x*sct); o[1]=(_Float16)(ta.y*sct);
        o[2]=(_Float16)(ta.z*sct); o[3]=(_Float16)(ta.w*sct);
        o[4]=(_Float16)(tb.x*sct); o[5]=(_Float16)(tb.y*sct);
        o[6]=(_Float16)(tb.z*sct); o[7]=(_Float16)(tb.w*sct);
        *(v8h*)(th + (size_t)row * DQ + l * 8) = o;
    }
}

// ---------------------------------------------------------------------------
// Async global -> LDS staging helpers (CDNA5 ASYNCcnt path, §10.7/§15.18.3)
// ---------------------------------------------------------------------------
__device__ __forceinline__ void async_b128_to_lds(unsigned lds_byte_addr,
                                                  const void* gaddr)
{
    asm volatile("global_load_async_to_lds_b128 %0, %1, off"
                 :: "v"(lds_byte_addr), "v"(gaddr)
                 : "memory");
}

__device__ __forceinline__ void wait_asynccnt_le2()
{
    asm volatile("s_wait_asynccnt 0x2" ::: "memory");
}
__device__ __forceinline__ void wait_asynccnt_0()
{
    asm volatile("s_wait_asynccnt 0x0" ::: "memory");
}

// Stage 64 rows of th (cols nb..nb+63, full K) into padded LDS tile:
// 512 x b128 chunks, 2 per thread, issued asynchronously.
__device__ __forceinline__ void stage_tile_async(const _Float16* __restrict__ th,
                                                 int nb, unsigned ldsTileBase,
                                                 int tid)
{
    const char* gbase = (const char*)(th + (size_t)nb * DQ);
#pragma unroll
    for (int it = 0; it < 2; ++it) {
        const int chunk = tid + it * 256;         // 0..511
        const int row = chunk >> 3;               // 8 x b128 per row
        const int c8  = chunk & 7;
        const unsigned lds = ldsTileBase + (unsigned)(row * (SB_STRIDE_H * 2) + c8 * 16);
        async_b128_to_lds(lds, gbase + (size_t)chunk * 16);
    }
}

// ---------------------------------------------------------------------------
// Kernel B: fused q @ t^T GEMM (f16 WMMA, f32 acc) + row-wise sum(exp(sim/T)).
// Grid: (BQ/128, NSPLIT). Wave w owns rows m0..m0+15; A frags live in VGPRs.
// B tiles double-buffered in LDS via async loads; next tile's HBM traffic
// overlaps the current tile's 32-WMMA compute phase. sched_group_barrier
// pipelines DS fragment reads against the WMMA chain.
// ---------------------------------------------------------------------------
__global__ __launch_bounds__(256) void infonce_gemm_lse(
    const _Float16* __restrict__ qh, const _Float16* __restrict__ th,
    float* __restrict__ partial, float* __restrict__ diag)
{
    __shared__ _Float16 sB[2 * TILE_HALVES];

    const int tid = threadIdx.x;
    const int w   = tid >> 5;
    const int l   = tid & 31;
    const int l16 = l & 15;
    const int hi  = l >> 4;                 // 0 = lanes 0-15, 1 = lanes 16-31
    const int m0  = blockIdx.x * 128 + w * 16;
    const int ns  = blockIdx.y;
    const int nbeg = ns * NCOLS_PER_SPLIT;
    const int nend = nbeg + NCOLS_PER_SPLIT;
    const bool have_diag = (m0 >= nbeg) && (m0 < nend);

    // Generic LDS pointer carries the LDS byte offset in addr[31:0] (§10.2).
    const unsigned ldsBase = (unsigned)(uintptr_t)(void*)&sB[0];

    // ---- Preload A fragments for this wave's 16 rows (K = 0..255) ----
    // A 16x32 f16 k order used here (contiguous, B matches): lane l16 = M; VGPR v pair = K = ks*32 + hi*16 + v*2
    const unsigned* qu = (const unsigned*)qh + (size_t)(m0 + l16) * (DQ / 2);
    Frag16 a[8];
#pragma unroll
    for (int ks = 0; ks < 8; ++ks) {
#pragma unroll
        for (int v = 0; v < 8; ++v) {
            const int k = ks * 32 + (hi << 4) + (v << 1);
            a[ks].u[v] = qu[k >> 1];
        }
    }

    float acc[8] = {0.f, 0.f, 0.f, 0.f, 0.f, 0.f, 0.f, 0.f};
    const unsigned* sBu = (const unsigned*)sB;

    // Prologue: kick off async staging of the first tile into buffer 0.
    stage_tile_async(th, nbeg, ldsBase, tid);

    int cur = 0;
    for (int nb = nbeg; nb < nend; nb += 64) {
        // All waves finished reading buffer cur^1 (previous compute) before
        // we overwrite it with the next tile.
        __syncthreads();
        const bool more = (nb + 64) < nend;
        if (more)
            stage_tile_async(th, nb + 64, ldsBase + (unsigned)((cur ^ 1) * TILE_BYTES), tid);
        if (more) wait_asynccnt_le2();   // current tile's 2 loads retired (in-order)
        else      wait_asynccnt_0();
        __syncthreads();                 // every thread's current-tile data landed

        const unsigned bufOff = (unsigned)(cur * (TILE_BYTES / 4));
#pragma unroll
        for (int sub = 0; sub < 4; ++sub) {
            // Batch-load all 8 B fragments, then run the WMMA chain; the
            // sched_group_barriers below pipeline DS reads against WMMAs.
            Frag16 b[8];
#pragma unroll
            for (int ks = 0; ks < 8; ++ks) {
#pragma unroll
                for (int v = 0; v < 8; ++v)
                    b[ks].u[v] = sBu[bufOff + (sub * 16 + l16) * SB_STRIDE_U32
                                     + ks * 16 + (hi << 3) + v];
            }
            v8f c = {0.f, 0.f, 0.f, 0.f, 0.f, 0.f, 0.f, 0.f};
#pragma unroll
            for (int ks = 0; ks < 8; ++ks)
                c = __builtin_amdgcn_wmma_f32_16x16x32_f16(
                        false, a[ks].h, false, b[ks].h, (short)0, c, false, false);

            // Pipeline shape: DS4 (frags 0-1), then [WMMA1, DS2] x 6, WMMA2.
            __builtin_amdgcn_sched_group_barrier(0x100, 4, 0);   // 4 DS reads
#pragma unroll
            for (int i = 0; i < 6; ++i) {
                __builtin_amdgcn_sched_group_barrier(0x008, 1, 0); // 1 WMMA
                __builtin_amdgcn_sched_group_barrier(0x100, 2, 0); // 2 DS reads
            }
            __builtin_amdgcn_sched_group_barrier(0x008, 2, 0);     // 2 WMMA

            // C layout: VGPR r, lane l -> M = r + 8*hi, N = l16.
            // exp(sim/T) == exp2(sim * log2e/T): single fused scale, raw v_exp_f32.
#pragma unroll
            for (int r = 0; r < 8; ++r)
                acc[r] += __builtin_amdgcn_exp2f(c[r] * EXP2_SCALE);
        }
        cur ^= 1;
    }

    // ---- Diagonal tile: recompute once, outside the hot loop. ----
    // B fragments for columns m0..m0+15 gathered straight from global th:
    // N = l16 -> row (m0+l16) of th; K = ks*32 + hi*16 + 2v -> u32 idx ks*16+hi*8+v.
    float dval = 0.0f;
    if (have_diag) {
        const unsigned* tu = (const unsigned*)th + (size_t)(m0 + l16) * (DQ / 2);
        v8f c = {0.f, 0.f, 0.f, 0.f, 0.f, 0.f, 0.f, 0.f};
#pragma unroll
        for (int ks = 0; ks < 8; ++ks) {
            Frag16 b;
#pragma unroll
            for (int v = 0; v < 8; ++v)
                b.u[v] = tu[ks * 16 + (hi << 3) + v];
            c = __builtin_amdgcn_wmma_f32_16x16x32_f16(
                    false, a[ks].h, false, b.h, (short)0, c, false, false);
        }
#pragma unroll
        for (int r = 0; r < 8; ++r)
            if (l16 == r + (hi << 3)) dval = c[r];
    }

    // Diag of row (m0 + 8*hi + l16) lives at lane (hi ? 24+l16 : l16).
    const int srcLane = hi ? (24 + l16) : l16;
    const float dg = __shfl(dval, srcLane, 32);

#pragma unroll
    for (int r = 0; r < 8; ++r) {
        float s = acc[r];
        s += __shfl_xor(s, 1, 16);
        s += __shfl_xor(s, 2, 16);
        s += __shfl_xor(s, 4, 16);
        s += __shfl_xor(s, 8, 16);
        if (l16 == r) {
            const int row = m0 + (hi << 3) + r;
            partial[ns * BQ + row] = s;
            if (have_diag) diag[row] = dg * INV_T;
        }
    }
}

// ---------------------------------------------------------------------------
// Kernel C: deterministic finalize. Sum NSPLIT partials per row in fixed
// order, loss_row = log(S) - diag, tree-reduce, mean -> d_out[0].
// ---------------------------------------------------------------------------
__global__ __launch_bounds__(256) void infonce_finalize(
    const float* __restrict__ partial, const float* __restrict__ diag,
    float* __restrict__ out)
{
    __shared__ float red[256];
    float local = 0.f;
    for (int m = threadIdx.x; m < BQ; m += 256) {
        float rs = partial[m];
#pragma unroll
        for (int p = 1; p < NSPLIT; ++p) rs += partial[p * BQ + m];
        local += logf(rs) - diag[m];
    }
    red[threadIdx.x] = local;
    __syncthreads();
    for (int s2 = 128; s2 > 0; s2 >>= 1) {
        if (threadIdx.x < s2) red[threadIdx.x] += red[threadIdx.x + s2];
        __syncthreads();
    }
    if (threadIdx.x == 0) out[0] = red[0] / (float)BQ;
}

// ---------------------------------------------------------------------------
extern "C" void kernel_launch(void* const* d_in, const int* in_sizes, int n_in,
                              void* d_out, int out_size, void* d_ws, size_t ws_size,
                              hipStream_t stream)
{
    const float* h = (const float*)d_in[0];
    const float* r = (const float*)d_in[1];
    const float* t = (const float*)d_in[2];

    char* ws = (char*)d_ws;
    _Float16* qh = (_Float16*)ws;                                  // 4 MB
    _Float16* th = (_Float16*)(ws + (size_t)BQ * DQ * 2);          // 4 MB
    float* partial = (float*)(ws + (size_t)BQ * DQ * 4);           // 256 KB
    float* diag    = partial + (size_t)NSPLIT * BQ;                // 32 KB

    infonce_normalize<<<BQ / 8, 256, 0, stream>>>(h, r, t, qh, th);
    dim3 gridB(BQ / 128, NSPLIT);
    infonce_gemm_lse<<<gridB, 256, 0, stream>>>(qh, th, partial, diag);
    infonce_finalize<<<1, 256, 0, stream>>>(partial, diag, (float*)d_out);
}
